// GAT_35107062677930
// MI455X (gfx1250) — matrix operand order repaired and checked by reading, not compiled, not run on hardware
//
#include <hip/hip_runtime.h>
#include <hip/hip_bf16.h>

typedef __attribute__((ext_vector_type(16))) __bf16 v16bf;
typedef __attribute__((ext_vector_type(8)))  float  v8f;

#define NEG_SLOPE 0.2f
#define CDIV(a, b) (((a) + (b) - 1) / (b))

// ---------------------------------------------------------------------------
// utilities
// ---------------------------------------------------------------------------
__global__ void k_fill_u32(unsigned* __restrict__ p, unsigned v, size_t n) {
  size_t i = (size_t)blockIdx.x * blockDim.x + threadIdx.x;
  size_t stride = (size_t)gridDim.x * blockDim.x;
  for (; i < n; i += stride) p[i] = v;
}

__global__ void k_cvt_bf16(const float* __restrict__ src, __bf16* __restrict__ dst, size_t n) {
  size_t i = (size_t)blockIdx.x * blockDim.x + threadIdx.x;
  size_t stride = (size_t)gridDim.x * blockDim.x;
  for (; i < n; i += stride) dst[i] = (__bf16)src[i];
}

// W[k, d] (f32, K x D row-major) -> Wt[d, k] (bf16, D x K row-major)
__global__ void k_pack_wt(const float* __restrict__ W, __bf16* __restrict__ Wt, int K, int D) {
  int i = blockIdx.x * blockDim.x + threadIdx.x;
  if (i >= K * D) return;
  int k = i / D, d = i - k * D;
  Wt[(size_t)d * K + k] = (__bf16)W[i];
}

// append self-loops to edge list; edge_index is [2, E] row-major
__global__ void k_build_edges(const int* __restrict__ ei, int* __restrict__ src,
                              int* __restrict__ dst, int E, int N) {
  int i = blockIdx.x * blockDim.x + threadIdx.x;
  if (i < E) {
    src[i] = ei[i];
    dst[i] = ei[E + i];
  } else if (i < E + N) {
    int v = i - E;
    src[i] = v;
    dst[i] = v;
  }
}

// ---------------------------------------------------------------------------
// bf16 WMMA GEMM: C[M,Nc] = A[M,K] * Bt[Nc,K]^T  (fp32 accumulate)
// One wave per 16x16 output tile, K-step 32 per v_wmma_f32_16x16x32_bf16.
// Requires M % 16 == 0, Nc % 16 == 0, K % 32 == 0 (true for this model).
// ---------------------------------------------------------------------------
__global__ void k_gemm_wmma_bf16(const __bf16* __restrict__ A,   // [M, K]
                                 const __bf16* __restrict__ Bt,  // [Nc, K]
                                 float* __restrict__ C,          // [M, Nc]
                                 int M, int K, int Nc) {
  const int lane  = threadIdx.x;                      // 0..31
  const int tileM = blockIdx.x * blockDim.y + threadIdx.y;
  const int tileN = blockIdx.y;
  if (tileM * 16 >= M) return;

  const int hi   = lane >> 4;                         // lane half: 0 or 1
  const int mrow = tileM * 16 + (lane & 15);          // A row handled by this lane
  const int ncol = tileN * 16 + (lane & 15);          // B column handled by this lane
  const __bf16* Ar = A  + (size_t)mrow * K;
  const __bf16* Br = Bt + (size_t)ncol * K;

  v8f acc = {};
  for (int k0 = 0; k0 < K; k0 += 32) {
    v16bf a, b;
    // A 16x32 bf16 fragment: VGPR v in 0..3 holds K pair (hi*16 + 2v, +1),
    // VGPRs 4..7 hold the same pattern shifted by K+8.
#pragma unroll
    for (int v = 0; v < 4; ++v) {
      int ka = k0 + hi * 16 + 2 * v;
      a[2 * v]         = Ar[ka];
      a[2 * v + 1]     = Ar[ka + 1];
      a[8 + 2 * v]     = Ar[ka + 8];
      a[8 + 2 * v + 1] = Ar[ka + 9];
    }
    // B 32x16 bf16 fragment: lanes 0-15 carry K=0..15, lanes 16-31 K=16..31,
    // VGPR v holds the packed pair (2v, 2v+1) for this lane's column.
#pragma unroll
    for (int v = 0; v < 8; ++v) {
      int kb = k0 + hi * 16 + 2 * v;
      b[2 * v]     = Br[kb];
      b[2 * v + 1] = Br[kb + 1];
    }
    acc = __builtin_amdgcn_wmma_f32_16x16x32_bf16(false, a, false, b,
                                                  (short)0, acc, false, false);
  }

  // C/D layout: VGPR r -> row tileM*16 + hi*8 + r, col = lane & 15
  const int rbase = tileM * 16 + hi * 8;
  const int cc    = tileN * 16 + (lane & 15);
#pragma unroll
  for (int r = 0; r < 8; ++r)
    C[(size_t)(rbase + r) * Nc + cc] = acc[r];
}

// ---------------------------------------------------------------------------
// per-node attention logits: oa[i] = h[i,:].a_src, ob[i] = h[i,:].a_dst
// ---------------------------------------------------------------------------
__global__ void k_rowdot2(const float* __restrict__ h, const float* __restrict__ va,
                          const float* __restrict__ vb, float* __restrict__ oa,
                          float* __restrict__ ob, int N, int D) {
  int node = blockIdx.x * (blockDim.x >> 5) + (threadIdx.x >> 5);
  int lane = threadIdx.x & 31;
  if (node >= N) return;
  const float* r = h + (size_t)node * D;
  float sa = 0.f, sb = 0.f;
  for (int f = lane; f < D; f += 32) {
    float x = r[f];
    sa += x * va[f];
    sb += x * vb[f];
  }
  for (int off = 16; off; off >>= 1) {
    sa += __shfl_xor(sa, off, 32);
    sb += __shfl_xor(sb, off, 32);
  }
  if (lane == 0) { oa[node] = sa; ob[node] = sb; }
}

// ---------------------------------------------------------------------------
// segment softmax over edges (grouped by dst)
// ---------------------------------------------------------------------------
__device__ __forceinline__ unsigned f2ord(float f) {
  unsigned u = __float_as_uint(f);
  return (u & 0x80000000u) ? ~u : (u | 0x80000000u);  // order-preserving map
}
__device__ __forceinline__ float ord2f(unsigned u) {
  return __uint_as_float((u & 0x80000000u) ? (u & 0x7fffffffu) : ~u);
}
__device__ __forceinline__ float leaky(float v) {
  return v > 0.f ? v : NEG_SLOPE * v;
}

__global__ void k_edge_max(const int* __restrict__ src, const int* __restrict__ dst,
                           const float* __restrict__ asrc, const float* __restrict__ adst,
                           unsigned* __restrict__ m_u, int Etot) {
  int e = blockIdx.x * blockDim.x + threadIdx.x;
  if (e >= Etot) return;
  int s = src[e], d = dst[e];
  atomicMax(&m_u[d], f2ord(leaky(asrc[s] + adst[d])));
}

__global__ void k_edge_exp_sum(const int* __restrict__ src, const int* __restrict__ dst,
                               const float* __restrict__ asrc, const float* __restrict__ adst,
                               const unsigned* __restrict__ m_u, float* __restrict__ denom,
                               float* __restrict__ ex, int Etot) {
  int e = blockIdx.x * blockDim.x + threadIdx.x;
  if (e >= Etot) return;
  int s = src[e], d = dst[e];
  float v = __expf(leaky(asrc[s] + adst[d]) - ord2f(m_u[d]));
  ex[e] = v;
  atomicAdd(&denom[d], v);
}

// out[dst,:] += (ex/denom[dst]) * h[src,:] ; one wave per edge (D multiple of 32)
__global__ void k_edge_aggregate(const int* __restrict__ src, const int* __restrict__ dst,
                                 const float* __restrict__ ex, const float* __restrict__ denom,
                                 const float* __restrict__ h, float* __restrict__ out,
                                 int Etot, int D) {
  int e    = blockIdx.x * (blockDim.x >> 5) + (threadIdx.x >> 5);
  int lane = threadIdx.x & 31;
  if (e >= Etot) return;
  int s = src[e], d = dst[e];
  float alpha = ex[e] / denom[d];
  const float* hs = h + (size_t)s * D;
  float* od = out + (size_t)d * D;
  for (int f = lane; f < D; f += 32)
    atomicAdd(&od[f], alpha * hs[f]);
}

// scalar-feature variant (D == 1), one thread per edge
__global__ void k_edge_aggregate1(const int* __restrict__ src, const int* __restrict__ dst,
                                  const float* __restrict__ ex, const float* __restrict__ denom,
                                  const float* __restrict__ h, float* __restrict__ out, int Etot) {
  int e = blockIdx.x * blockDim.x + threadIdx.x;
  if (e >= Etot) return;
  int s = src[e], d = dst[e];
  atomicAdd(&out[d], (ex[e] / denom[d]) * h[s]);
}

// ---------------------------------------------------------------------------
// epilogues: +bias, ELU, optional bf16 repack for the next WMMA GEMM
// ---------------------------------------------------------------------------
__device__ __forceinline__ float elu(float v) { return v > 0.f ? v : (__expf(v) - 1.f); }

__global__ void k_bias_elu_bf16(const float* __restrict__ acc, const float* __restrict__ bias,
                                __bf16* __restrict__ out, size_t total, int D) {
  size_t i = (size_t)blockIdx.x * blockDim.x + threadIdx.x;
  if (i >= total) return;
  out[i] = (__bf16)elu(acc[i] + bias[i % D]);
}

__global__ void k_bias_elu_f32(const float* __restrict__ acc, const float* __restrict__ bias,
                               float* __restrict__ out, size_t total, int D) {
  size_t i = (size_t)blockIdx.x * blockDim.x + threadIdx.x;
  if (i >= total) return;
  out[i] = elu(acc[i] + bias[i % D]);
}

__global__ void k_add_scalar(float* __restrict__ p, const float* __restrict__ b, int n) {
  int i = blockIdx.x * blockDim.x + threadIdx.x;
  if (i < n) p[i] += b[0];
}

// layer 3: h3 = act2[64] . W3[64,1]; also alpha_src/alpha_dst scalars. One wave/node.
__global__ void k_layer3_h(const float* __restrict__ act2, const float* __restrict__ W3,
                           const float* __restrict__ as3, const float* __restrict__ ad3,
                           float* __restrict__ h3, float* __restrict__ asrc,
                           float* __restrict__ adst, int N) {
  int node = blockIdx.x * (blockDim.x >> 5) + (threadIdx.x >> 5);
  int lane = threadIdx.x & 31;
  if (node >= N) return;
  const float* r = act2 + (size_t)node * 64;
  float s = r[lane] * W3[lane] + r[lane + 32] * W3[lane + 32];
  for (int off = 16; off; off >>= 1) s += __shfl_xor(s, off, 32);
  if (lane == 0) {
    h3[node]   = s;
    asrc[node] = s * as3[0];
    adst[node] = s * ad3[0];
  }
}

// ---------------------------------------------------------------------------
// launch
// ---------------------------------------------------------------------------
extern "C" void kernel_launch(void* const* d_in, const int* in_sizes, int n_in,
                              void* d_out, int out_size, void* d_ws, size_t ws_size,
                              hipStream_t stream) {
  (void)n_in; (void)out_size; (void)ws_size;

  const int C  = 128, D1 = 256, D2 = 64;
  const int N    = in_sizes[0] / C;      // 50000
  const int E    = in_sizes[1] / 2;      // 800000
  const int Etot = E + N;                // + self loops

  const float* x   = (const float*)d_in[0];
  const int*   ei  = (const int*)d_in[1];
  const float* W1  = (const float*)d_in[2];
  const float* as1 = (const float*)d_in[3];
  const float* ad1 = (const float*)d_in[4];
  const float* b1  = (const float*)d_in[5];
  const float* W2  = (const float*)d_in[6];
  const float* as2 = (const float*)d_in[7];
  const float* ad2 = (const float*)d_in[8];
  const float* b2  = (const float*)d_in[9];
  const float* W3  = (const float*)d_in[10];
  const float* as3 = (const float*)d_in[11];
  const float* ad3 = (const float*)d_in[12];
  const float* b3  = (const float*)d_in[13];
  float* out = (float*)d_out;

  // ---- bump allocator over d_ws (256B aligned) with time-based aliasing ----
  char* base = (char*)d_ws;
  size_t off = 0;
  auto alloc = [&](size_t bytes) -> char* {
    char* p = base + off;
    off += (bytes + 255) & ~(size_t)255;
    return p;
  };
  int*      src2  = (int*)alloc((size_t)Etot * 4);
  int*      dst2  = (int*)alloc((size_t)Etot * 4);
  float*    exbuf = (float*)alloc((size_t)Etot * 4);
  __bf16*   Wt1   = (__bf16*)alloc((size_t)D1 * C * 2);
  __bf16*   Wt2   = (__bf16*)alloc((size_t)D2 * D1 * 2);
  float*    asrc  = (float*)alloc((size_t)N * 4);
  float*    adst  = (float*)alloc((size_t)N * 4);
  unsigned* m_u   = (unsigned*)alloc((size_t)N * 4);
  float*    denom = (float*)alloc((size_t)N * 4);
  float*    h3    = (float*)alloc((size_t)N * 4);
  // region A: xb (gemm1 input) then reused as h1b (gemm2 input)
  char* regA = alloc((size_t)N * D1 * 2);
  __bf16* xb  = (__bf16*)regA;
  __bf16* h1b = (__bf16*)regA;
  // region B: h1 (layer1), later {h2, out2, act2} (layer2/3)
  char* regB = alloc((size_t)N * D1 * 4);
  float* h1   = (float*)regB;
  float* h2   = (float*)regB;
  float* out2 = (float*)(regB + (size_t)N * D2 * 4);
  float* act2 = (float*)(regB + 2 * (size_t)N * D2 * 4);
  // region C: out1 accumulator
  float* out1 = (float*)alloc((size_t)N * D1 * 4);

  const int T = 256;
  const dim3 gemmBlk(32, 4);                 // 4 waves per block, 1 tile per wave
  const int  mtiles = CDIV(N, 16);

  // ---- shared preprocessing ----
  k_build_edges<<<CDIV(Etot, T), T, 0, stream>>>(ei, src2, dst2, E, N);
  k_cvt_bf16<<<1024, T, 0, stream>>>(x, xb, (size_t)N * C);
  k_pack_wt<<<CDIV(C * D1, T), T, 0, stream>>>(W1, Wt1, C, D1);
  k_pack_wt<<<CDIV(D1 * D2, T), T, 0, stream>>>(W2, Wt2, D1, D2);

  // ================= layer 1 =================
  k_gemm_wmma_bf16<<<dim3(CDIV(mtiles, 4), D1 / 16), gemmBlk, 0, stream>>>(xb, Wt1, h1, N, C, D1);
  k_rowdot2<<<CDIV(N, 8), T, 0, stream>>>(h1, as1, ad1, asrc, adst, N, D1);
  k_fill_u32<<<256, T, 0, stream>>>(m_u, 0u, (size_t)N);
  k_fill_u32<<<256, T, 0, stream>>>((unsigned*)denom, 0u, (size_t)N);
  k_fill_u32<<<1024, T, 0, stream>>>((unsigned*)out1, 0u, (size_t)N * D1);
  k_edge_max<<<CDIV(Etot, T), T, 0, stream>>>(src2, dst2, asrc, adst, m_u, Etot);
  k_edge_exp_sum<<<CDIV(Etot, T), T, 0, stream>>>(src2, dst2, asrc, adst, m_u, denom, exbuf, Etot);
  k_edge_aggregate<<<CDIV(Etot, 8), T, 0, stream>>>(src2, dst2, exbuf, denom, h1, out1, Etot, D1);
  k_bias_elu_bf16<<<CDIV((size_t)N * D1, T), T, 0, stream>>>(out1, b1, h1b, (size_t)N * D1, D1);

  // ================= layer 2 =================
  k_gemm_wmma_bf16<<<dim3(CDIV(mtiles, 4), D2 / 16), gemmBlk, 0, stream>>>(h1b, Wt2, h2, N, D1, D2);
  k_rowdot2<<<CDIV(N, 8), T, 0, stream>>>(h2, as2, ad2, asrc, adst, N, D2);
  k_fill_u32<<<256, T, 0, stream>>>(m_u, 0u, (size_t)N);
  k_fill_u32<<<256, T, 0, stream>>>((unsigned*)denom, 0u, (size_t)N);
  k_fill_u32<<<1024, T, 0, stream>>>((unsigned*)out2, 0u, (size_t)N * D2);
  k_edge_max<<<CDIV(Etot, T), T, 0, stream>>>(src2, dst2, asrc, adst, m_u, Etot);
  k_edge_exp_sum<<<CDIV(Etot, T), T, 0, stream>>>(src2, dst2, asrc, adst, m_u, denom, exbuf, Etot);
  k_edge_aggregate<<<CDIV(Etot, 8), T, 0, stream>>>(src2, dst2, exbuf, denom, h2, out2, Etot, D2);
  k_bias_elu_f32<<<CDIV((size_t)N * D2, T), T, 0, stream>>>(out2, b2, act2, (size_t)N * D2, D2);

  // ================= layer 3 =================
  k_layer3_h<<<CDIV(N, 8), T, 0, stream>>>(act2, W3, as3, ad3, h3, asrc, adst, N);
  k_fill_u32<<<256, T, 0, stream>>>(m_u, 0u, (size_t)N);
  k_fill_u32<<<256, T, 0, stream>>>((unsigned*)denom, 0u, (size_t)N);
  k_fill_u32<<<256, T, 0, stream>>>((unsigned*)out, 0u, (size_t)N);
  k_edge_max<<<CDIV(Etot, T), T, 0, stream>>>(src2, dst2, asrc, adst, m_u, Etot);
  k_edge_exp_sum<<<CDIV(Etot, T), T, 0, stream>>>(src2, dst2, asrc, adst, m_u, denom, exbuf, Etot);
  k_edge_aggregate1<<<CDIV(Etot, T), T, 0, stream>>>(src2, dst2, exbuf, denom, h3, out, Etot);
  k_add_scalar<<<CDIV(N, T), T, 0, stream>>>(out, b3, N);
}
